// ViewGenerator_16209206575453
// MI455X (gfx1250) — matrix operand order repaired and checked by reading, not compiled, not run on hardware
//
#include <hip/hip_runtime.h>
#include <math.h>

// ---------------------------------------------------------------------------
// ViewGenerator for MI455X (gfx1250, wave32, WMMA)
//
// MLP heads: V_WMMA_F32_16X16X32_F16 with f16 hi/lo split of f32 operands
// (AhBh + AlBh + AhBl, f32 accumulate) -> ~f32 accuracy at f16 matrix rate.
// BN params staged into LDS via GLOBAL_LOAD_ASYNC_TO_LDS (ASYNCcnt).
// Dominant cost is the streaming rotation pass (~184 MB @ 23.3 TB/s HBM)
// -> float4 non-temporal loads/stores.
// ---------------------------------------------------------------------------

typedef __attribute__((ext_vector_type(16))) _Float16 v16h;
typedef __attribute__((ext_vector_type(8)))  float    v8f;
typedef __attribute__((ext_vector_type(4)))  float    f4;

#define N_BATCH 512
#define HID     256
#define NPTS    15000            // 300*25*2 points per (n, channel)
#define NP4     (NPTS / 4)       // 3750 float4 per (n, channel)

#if __has_builtin(__builtin_amdgcn_global_load_async_to_lds_b32)
#define HAVE_ASYNC_LDS 1
typedef __attribute__((address_space(1))) int gas_int;
typedef __attribute__((address_space(3))) int las_int;
__device__ __forceinline__ void async_g2l_b32(const float* g, float* l) {
  __builtin_amdgcn_global_load_async_to_lds_b32(
      (gas_int*)(g), (las_int*)(l), 0, 0);
}
#endif

__device__ __forceinline__ void f32_split(float f, _Float16& h, _Float16& l) {
  h = (_Float16)f;
  l = (_Float16)(f - (float)h);
}

__device__ __forceinline__ v8f wmma_acc(v16h a, v16h b, v8f c) {
  // 8 args: (neg_a, A, neg_b, B, c_mod, C, reuse_a, reuse_b)
  return __builtin_amdgcn_wmma_f32_16x16x32_f16(false, a, false, b, (short)0, c,
                                                false, false);
}

__device__ __forceinline__ float lrelu(float t) { return t >= 0.0f ? t : 0.01f * t; }

// ---------------------------------------------------------------------------
// Layer 0 GEMM: Y = concat(x, noise) @ W^T + b   (512x512 @ 512x256)
// 8 waves/block, each wave one 16x16 tile; block covers 32 rows x 64 cols.
// Fragment addressing per CDNA5 ISA 7.12.2 (16-bit A 16x32 / B 32x16 / f32 C).
// ---------------------------------------------------------------------------
__global__ __launch_bounds__(256) void gemm_first_k(
    const float* __restrict__ x, const float* __restrict__ noise,
    const float* __restrict__ W, const float* __restrict__ bias,
    float* __restrict__ Y)
{
  const int K    = 2 * HID;
  const int tid  = threadIdx.x;
  const int wave = tid >> 5;
  const int lane = tid & 31;
  const int l    = lane & 15;
  const int hb   = lane >> 4;

  const int rowBase = blockIdx.y * 32 + (wave >> 2) * 16;
  const int colBase = blockIdx.x * 64 + (wave & 3) * 16;

  const int arow = rowBase + l;                     // A row held by this lane
  const float* xr = x     + (size_t)arow * HID;
  const float* nr = noise + (size_t)arow * HID;
  const float* wr = W + (size_t)(colBase + l) * K;  // B column = W row

  v8f acc = {};
#pragma unroll 2
  for (int kb = 0; kb < K; kb += 32) {
    // a K-chunk of 32 never straddles the x/noise boundary (256 % 32 == 0)
    const float* srow = (kb < HID) ? xr : (nr - HID);
    v16h ah, al, bh, bl;
#pragma unroll
    for (int v = 0; v < 8; ++v) {
      // A: lanes 0-15 hold K pairs at {0,16}+2v; lanes 16-31 at +8/+24
      const int ka = kb + ((v < 4) ? 0 : 16) + hb * 8 + 2 * (v & 3);
      const float2 av = *(const float2*)(srow + ka);       // global_load_b64
      _Float16 h0, h1, l0, l1;
      f32_split(av.x, h0, l0);
      f32_split(av.y, h1, l1);
      ah[2 * v] = h0; ah[2 * v + 1] = h1;
      al[2 * v] = l0; al[2 * v + 1] = l1;
      // B: same K order as A (lanes 0-15 K pairs at {0,16}+2v; lanes 16-31 at +8/+24)
      const float2 bv = *(const float2*)(wr + kb + ((v < 4) ? 0 : 16) + hb * 8 + 2 * (v & 3));
      f32_split(bv.x, h0, l0);
      f32_split(bv.y, h1, l1);
      bh[2 * v] = h0; bh[2 * v + 1] = h1;
      bl[2 * v] = l0; bl[2 * v + 1] = l1;
    }
    acc = wmma_acc(ah, bh, acc);
    acc = wmma_acc(al, bh, acc);
    acc = wmma_acc(ah, bl, acc);
  }
  const float bv = bias[colBase + l];
#pragma unroll
  for (int j = 0; j < 8; ++j)                        // C: VGPR j -> row hb*8+j
    Y[(size_t)(rowBase + hb * 8 + j) * HID + (colBase + l)] = acc[j] + bv;
}

// ---------------------------------------------------------------------------
// Hidden-layer GEMM with previous layer's BN + LeakyReLU fused into the
// A-operand load:  Y = lrelu(Yp*scale + shift) @ W^T + b
// BN params staged through LDS (async global->LDS when available).
// ---------------------------------------------------------------------------
__global__ __launch_bounds__(256) void gemm_bn_k(
    const float* __restrict__ Yp,
    const float* __restrict__ mean, const float* __restrict__ rstd,
    const float* __restrict__ gam,  const float* __restrict__ bet,
    const float* __restrict__ W,    const float* __restrict__ bias,
    float* __restrict__ Y)
{
  __shared__ float praw[4 * HID];
  __shared__ float ssc[HID], ssh[HID];
  const int tid = threadIdx.x;
#ifdef HAVE_ASYNC_LDS
  async_g2l_b32(mean + tid, &praw[tid]);
  async_g2l_b32(rstd + tid, &praw[HID + tid]);
  async_g2l_b32(gam  + tid, &praw[2 * HID + tid]);
  async_g2l_b32(bet  + tid, &praw[3 * HID + tid]);
#if __has_builtin(__builtin_amdgcn_s_wait_asynccnt)
  __builtin_amdgcn_s_wait_asynccnt(0);
#else
  asm volatile("s_wait_asynccnt 0" ::: "memory");
#endif
#else
  praw[tid]           = mean[tid];
  praw[HID + tid]     = rstd[tid];
  praw[2 * HID + tid] = gam[tid];
  praw[3 * HID + tid] = bet[tid];
#endif
  __syncthreads();
  {
    const float sc = praw[HID + tid] * praw[2 * HID + tid];   // rstd*g
    ssc[tid] = sc;
    ssh[tid] = praw[3 * HID + tid] - praw[tid] * sc;          // beta - mean*sc
  }
  __syncthreads();

  const int wave = tid >> 5;
  const int lane = tid & 31;
  const int l    = lane & 15;
  const int hb   = lane >> 4;

  const int rowBase = blockIdx.y * 32 + (wave >> 2) * 16;
  const int colBase = blockIdx.x * 64 + (wave & 3) * 16;

  const float* yr = Yp + (size_t)(rowBase + l) * HID;
  const float* wr = W + (size_t)(colBase + l) * HID;

  v8f acc = {};
#pragma unroll 2
  for (int kb = 0; kb < HID; kb += 32) {
    v16h ah, al, bh, bl;
#pragma unroll
    for (int v = 0; v < 8; ++v) {
      const int ka = kb + ((v < 4) ? 0 : 16) + hb * 8 + 2 * (v & 3);
      const float2 yv = *(const float2*)(yr + ka);           // global_load_b64
      const float a0 = lrelu(fmaf(yv.x, ssc[ka],     ssh[ka]));
      const float a1 = lrelu(fmaf(yv.y, ssc[ka + 1], ssh[ka + 1]));
      _Float16 h0, h1, l0, l1;
      f32_split(a0, h0, l0);
      f32_split(a1, h1, l1);
      ah[2 * v] = h0; ah[2 * v + 1] = h1;
      al[2 * v] = l0; al[2 * v + 1] = l1;
      const float2 bv = *(const float2*)(wr + kb + ((v < 4) ? 0 : 16) + hb * 8 + 2 * (v & 3));
      f32_split(bv.x, h0, l0);
      f32_split(bv.y, h1, l1);
      bh[2 * v] = h0; bh[2 * v + 1] = h1;
      bl[2 * v] = l0; bl[2 * v + 1] = l1;
    }
    acc = wmma_acc(ah, bh, acc);
    acc = wmma_acc(al, bh, acc);
    acc = wmma_acc(ah, bl, acc);
  }
  const float bv = bias[colBase + l];
#pragma unroll
  for (int j = 0; j < 8; ++j)
    Y[(size_t)(rowBase + hb * 8 + j) * HID + (colBase + l)] = acc[j] + bv;
}

// ---------------------------------------------------------------------------
// Batch-norm statistics over the batch dimension: one block per feature.
// ---------------------------------------------------------------------------
__global__ __launch_bounds__(256) void bn_stats_k(
    const float* __restrict__ Y, float* __restrict__ mean, float* __restrict__ rstd)
{
  __shared__ float s1[256], s2[256];
  const int col = blockIdx.x;
  const int tid = threadIdx.x;
  const float a = Y[(size_t)tid * HID + col];
  const float b = Y[(size_t)(tid + 256) * HID + col];
  s1[tid] = a + b;
  s2[tid] = a * a + b * b;
  __syncthreads();
  for (int off = 128; off > 0; off >>= 1) {
    if (tid < off) { s1[tid] += s1[tid + off]; s2[tid] += s2[tid + off]; }
    __syncthreads();
  }
  if (tid == 0) {
    const float m = s1[0] * (1.0f / N_BATCH);
    const float v = s2[0] * (1.0f / N_BATCH) - m * m;
    mean[col] = m;
    rstd[col] = rsqrtf(v + 1e-5f);
  }
}

// ---------------------------------------------------------------------------
// Final projection + tanh*theta: one wave per sample.
// ---------------------------------------------------------------------------
__global__ __launch_bounds__(32) void head_out_k(
    const float* __restrict__ Y,
    const float* __restrict__ mean, const float* __restrict__ rstd,
    const float* __restrict__ gam,  const float* __restrict__ bet,
    const float* __restrict__ w2,   const float* __restrict__ b2,
    float theta, float* __restrict__ r_out)
{
  const int n    = blockIdx.x;
  const int lane = threadIdx.x;
  float a0 = 0.f, a1 = 0.f, a2 = 0.f;
  for (int k = lane; k < HID; k += 32) {
    float h = (Y[(size_t)n * HID + k] - mean[k]) * rstd[k] * gam[k] + bet[k];
    h = lrelu(h);
    a0 += h * w2[k];
    a1 += h * w2[HID + k];
    a2 += h * w2[2 * HID + k];
  }
  for (int off = 16; off > 0; off >>= 1) {
    a0 += __shfl_down(a0, off, 32);
    a1 += __shfl_down(a1, off, 32);
    a2 += __shfl_down(a2, off, 32);
  }
  if (lane == 0) {
    r_out[n * 3 + 0] = tanhf(a0 + b2[0]) * theta;
    r_out[n * 3 + 1] = tanhf(a1 + b2[1]) * theta;
    r_out[n * 3 + 2] = tanhf(a2 + b2[2]) * theta;
  }
}

// ---------------------------------------------------------------------------
// Angle-axis -> rotation matrix (one thread per sample).
// ---------------------------------------------------------------------------
__global__ __launch_bounds__(256) void rotmat_k(
    const float* __restrict__ r, float* __restrict__ R)
{
  const int n = blockIdx.x * blockDim.x + threadIdx.x;
  if (n >= N_BATCH) return;
  const float rx = r[n * 3 + 0], ry = r[n * 3 + 1], rz = r[n * 3 + 2];
  const float th2 = rx * rx + ry * ry + rz * rz;
  float* Rn = R + n * 9;
  if (th2 > 1e-6f) {
    const float th = sqrtf(th2);
    const float inv = 1.0f / th;
    const float wx = rx * inv, wy = ry * inv, wz = rz * inv;
    const float c = cosf(th), s = sinf(th), oc = 1.0f - c;
    Rn[0] = c + wx * wx * oc;      Rn[1] = wx * wy * oc - wz * s; Rn[2] = wx * wz * oc + wy * s;
    Rn[3] = wy * wx * oc + wz * s; Rn[4] = c + wy * wy * oc;      Rn[5] = wy * wz * oc - wx * s;
    Rn[6] = wz * wx * oc - wy * s; Rn[7] = wz * wy * oc + wx * s; Rn[8] = c + wz * wz * oc;
  } else {
    Rn[0] = 1.f;  Rn[1] = -rz; Rn[2] = ry;
    Rn[3] = rz;   Rn[4] = 1.f; Rn[5] = -rx;
    Rn[6] = -ry;  Rn[7] = rx;  Rn[8] = 1.f;
  }
}

// ---------------------------------------------------------------------------
// Bandwidth-dominant pass: out[n,i,:] = sum_j R[n,i,j]*aug[n,j,:] + t[n,i]
// float4 non-temporal streaming (read-once/write-once; don't thrash L2).
// ---------------------------------------------------------------------------
__global__ __launch_bounds__(256) void rotate_apply_k(
    const float* __restrict__ aug, const float* __restrict__ R,
    const float* __restrict__ t,   float* __restrict__ out)
{
  const int n  = blockIdx.y;
  const int p4 = blockIdx.x * 256 + threadIdx.x;
  if (p4 >= NP4) return;

  const float* Rn = R + n * 9;
  const float* tn = t + n * 3;
  __builtin_prefetch(Rn, 0, 0);   // global_prefetch_b8

  const float* A = aug + (size_t)n * 3 * NPTS;
  const f4 v0 = __builtin_nontemporal_load(((const f4*)(A + 0 * NPTS)) + p4);
  const f4 v1 = __builtin_nontemporal_load(((const f4*)(A + 1 * NPTS)) + p4);
  const f4 v2 = __builtin_nontemporal_load(((const f4*)(A + 2 * NPTS)) + p4);

  float* O = out + (size_t)n * 3 * NPTS;
#pragma unroll
  for (int i = 0; i < 3; ++i) {
    const float r0 = Rn[i * 3 + 0], r1 = Rn[i * 3 + 1], r2 = Rn[i * 3 + 2];
    const float ti = tn[i];
    f4 o;
    o.x = r0 * v0.x + r1 * v1.x + r2 * v2.x + ti;
    o.y = r0 * v0.y + r1 * v1.y + r2 * v2.y + ti;
    o.z = r0 * v0.z + r1 * v1.z + r2 * v2.z + ti;
    o.w = r0 * v0.w + r1 * v1.w + r2 * v2.w + ti;
    __builtin_nontemporal_store(o, ((f4*)(O + i * NPTS)) + p4);
  }
}

// ---------------------------------------------------------------------------
// Host side
// ---------------------------------------------------------------------------
static void run_head(const float* x, const float* noise,
                     const float* w1w, const float* w1b,
                     const float* g0, const float* b0,
                     const float* stw, const float* stb,
                     const float* stg, const float* stbeta,
                     const float* w2w, const float* w2b,
                     float theta, float* head_r,
                     float* Y0, float* Y1, float* mean, float* rstd,
                     hipStream_t stream)
{
  const dim3 gG(4, 16), bG(256);
  gemm_first_k<<<gG, bG, 0, stream>>>(x, noise, w1w, w1b, Y0);
  bn_stats_k<<<HID, 256, 0, stream>>>(Y0, mean, rstd);

  const float* g  = g0;
  const float* bt = b0;
  float* Yp = Y0;
  float* Yc = Y1;
  for (int li = 0; li < 4; ++li) {
    gemm_bn_k<<<gG, bG, 0, stream>>>(Yp, mean, rstd, g, bt,
                                     stw + (size_t)li * HID * HID,
                                     stb + (size_t)li * HID, Yc);
    bn_stats_k<<<HID, 256, 0, stream>>>(Yc, mean, rstd);
    g  = stg    + (size_t)li * HID;
    bt = stbeta + (size_t)li * HID;
    float* tmp = Yp; Yp = Yc; Yc = tmp;
  }
  head_out_k<<<N_BATCH, 32, 0, stream>>>(Yp, mean, rstd, g, bt, w2w, w2b, theta, head_r);
}

extern "C" void kernel_launch(void* const* d_in, const int* in_sizes, int n_in,
                              void* d_out, int out_size, void* d_ws, size_t ws_size,
                              hipStream_t stream)
{
  (void)in_sizes; (void)n_in; (void)out_size; (void)ws_size;

  const float* x       = (const float*)d_in[0];
  const float* noise_R = (const float*)d_in[1];
  const float* noise_T = (const float*)d_in[2];
  const float* aug     = (const float*)d_in[3];
  const float* w1R_w   = (const float*)d_in[4];
  const float* w1R_b   = (const float*)d_in[5];
  const float* bnR_g   = (const float*)d_in[6];
  const float* bnR_b   = (const float*)d_in[7];
  const float* stR_w   = (const float*)d_in[8];
  const float* stR_b   = (const float*)d_in[9];
  const float* stR_g   = (const float*)d_in[10];
  const float* stR_be  = (const float*)d_in[11];
  const float* w2R_w   = (const float*)d_in[12];
  const float* w2R_b   = (const float*)d_in[13];
  const float* w1T_w   = (const float*)d_in[14];
  const float* w1T_b   = (const float*)d_in[15];
  const float* bnT_g   = (const float*)d_in[16];
  const float* bnT_b   = (const float*)d_in[17];
  const float* stT_w   = (const float*)d_in[18];
  const float* stT_b   = (const float*)d_in[19];
  const float* stT_g   = (const float*)d_in[20];
  const float* stT_be  = (const float*)d_in[21];
  const float* w2T_w   = (const float*)d_in[22];
  const float* w2T_b   = (const float*)d_in[23];

  float* out   = (float*)d_out;                       // (512,3,300,25,2)
  float* r_out = out + (size_t)N_BATCH * 3 * NPTS;    // (512,3)
  float* t_out = r_out + (size_t)N_BATCH * 3;         // (512,3)

  // Workspace layout
  float* Y0   = (float*)d_ws;                         // 512*256
  float* Y1   = Y0 + (size_t)N_BATCH * HID;           // 512*256
  float* mean = Y1 + (size_t)N_BATCH * HID;           // 256
  float* rstd = mean + HID;                           // 256
  float* Rm   = rstd + HID;                           // 512*9

  // R head -> r, rotation matrices
  run_head(x, noise_R, w1R_w, w1R_b, bnR_g, bnR_b, stR_w, stR_b, stR_g, stR_be,
           w2R_w, w2R_b, 0.3f, r_out, Y0, Y1, mean, rstd, stream);
  rotmat_k<<<(N_BATCH + 255) / 256, 256, 0, stream>>>(r_out, Rm);

  // T head -> t
  run_head(x, noise_T, w1T_w, w1T_b, bnT_g, bnT_b, stT_w, stT_b, stT_g, stT_be,
           w2T_w, w2T_b, 0.5f, t_out, Y0, Y1, mean, rstd, stream);

  // Streaming rotation + translation (HBM-bound)
  rotate_apply_k<<<dim3((NP4 + 255) / 256, N_BATCH), 256, 0, stream>>>(aug, Rm, t_out, out);
}
